// Denoiser_59828894433753
// MI455X (gfx1250) — matrix operand order repaired and checked by reading, not compiled, not run on hardware
//
#include <hip/hip_runtime.h>
#include <hip/hip_bf16.h>
#include <math.h>

// ---------------- problem constants ----------------
constexpr int Bn = 64;          // batch
constexpr int Tn = 256;         // time
constexpr int En = 256;         // embedding dim
constexpr int Hn = 256;         // hidden
constexpr int KW = En + Hn;     // 512 rows of Wk
constexpr int G4 = 4 * Hn;      // 1024 gate columns
constexpr int NROWS = Bn * Tn;  // 16384 rows of X / Zx
constexpr int NTILES_N = G4 / 16;   // 64 column tiles
constexpr int NKBLK = KW / 32;      // 16 K-blocks of 32

typedef __attribute__((ext_vector_type(16))) __bf16 bf16x16;
typedef __attribute__((ext_vector_type(8)))  __bf16 bf16x8;
typedef __attribute__((ext_vector_type(8)))  float  f32x8;

__device__ __forceinline__ float sigmoidf_(float x) {
    return 1.0f / (1.0f + __expf(-x));
}

// A-matrix fragment (16x32 bf16, row-major source, lda elements per row)
// lane L: M = L&15 ; K-half = (L>=16)?8:0 ; per-lane data = two contiguous
// 8-element runs -> compiler merges to 2x global_load_b128.
__device__ __forceinline__ bf16x16 load_a_frag(const __bf16* A, int lda,
                                               int rowBase, int kk, int lane) {
    const int m     = lane & 15;
    const int khalf = (lane >> 4) << 3;
    const __bf16* p = A + (size_t)(rowBase + m) * lda;
    bf16x16 a;
#pragma unroll
    for (int v = 0; v < 8; ++v) {
        int k0 = kk + ((v >> 2) << 4) + khalf + ((v & 3) << 1);
        a[2 * v]     = p[k0];
        a[2 * v + 1] = p[k0 + 1];
    }
    return a;
}

// B-matrix fragment from pre-swizzled weights:
// Wpk[dir][kblock][ntile][lane][16] -> single 32B vector load per lane
__device__ __forceinline__ bf16x16 load_b_packed(const __bf16* Wpk_dir,
                                                 int kblock, int ntile, int lane) {
    const __bf16* p = Wpk_dir +
        ((((size_t)kblock * NTILES_N + ntile) * 32 + lane) << 4);
    return *(const bf16x16*)p;
}

// ---------------- kernel 1: weights -> bf16, fragment-swizzled ----------------
// Wpk layout: [dir][kblock(16)][ntile(64)][lane(32)][16]
__global__ void k_cvt_weights(const float* __restrict__ Wkf,
                              const float* __restrict__ Wkb,
                              __bf16* __restrict__ Wpk) {
    int idx = blockIdx.x * blockDim.x + threadIdx.x;   // 2*512*1024
    if (idx >= 2 * KW * G4) return;
    int j      = idx & 15;
    int lane   = (idx >> 4) & 31;
    int ntile  = (idx >> 9) & 63;
    int kblock = (idx >> 15) & 15;
    int dir    = idx >> 19;
    int k   = kblock * 32 + ((j >> 3) << 4) + ((lane >> 4) << 3) + (j & 7);
    int col = ntile * 16 + (lane & 15);
    const float* src = dir ? Wkb : Wkf;
    Wpk[idx] = (__bf16)src[(size_t)k * G4 + col];
}

// ---------------- kernel 2: embedding gather -> bf16 X (time-major) ----------------
// X_bf layout: [t*B + b][E]
__global__ void k_embed(const int* __restrict__ tokens,
                        const float* __restrict__ emb,
                        __bf16* __restrict__ X_bf) {
    int idx = blockIdx.x * blockDim.x + threadIdx.x;   // 16384*256
    if (idx >= NROWS * En) return;
    int row = idx >> 8;        // t*64 + b
    int e   = idx & 255;
    int t   = row >> 6;
    int b   = row & 63;
    int tok = tokens[b * Tn + t];
    X_bf[idx] = (__bf16)emb[(size_t)tok * En + e];
}

// ---------------- kernel 3: zero LSTM state ----------------
__global__ void k_init_state(float* __restrict__ c, __bf16* __restrict__ hbf) {
    int idx = blockIdx.x * blockDim.x + threadIdx.x;   // 2*64*256
    if (idx >= 2 * Bn * Hn) return;
    c[idx]   = 0.0f;
    hbf[idx] = (__bf16)0.0f;
}

// ---------------- kernel 4: Zx = X @ Wx + bias (both dirs) ----------------
// one wave -> one 16x16 tile; output stored in C-fragment layout:
// Zx[dir][mtile(1024)][ntile(64)][lane(32)][8] bf16
__global__ void k_zx_gemm(const __bf16* __restrict__ X_bf,
                          const __bf16* __restrict__ Wpk,
                          const float* __restrict__ bias_f,
                          const float* __restrict__ bias_b,
                          __bf16* __restrict__ Zx) {
    const int lane = threadIdx.x & 31;
    const int gid  = blockIdx.x * (blockDim.x >> 5) + (threadIdx.x >> 5);
    // tiles: 2 dirs * 1024 Mtiles * 64 Ntiles = 131072
    const int dir = gid >> 16;
    const int rem = gid & 65535;
    const int mt  = rem >> 6;
    const int nt  = rem & 63;

    const __bf16* Wpk_dir = Wpk + (size_t)dir * KW * G4;   // kblocks 0..7 = Wx
    const float*  bias    = dir ? bias_b : bias_f;
    const int rowBase = mt * 16;
    const int col     = nt * 16 + (lane & 15);

    f32x8 acc = {};
#pragma unroll
    for (int kb = 0; kb < En / 32; ++kb) {
        bf16x16 a = load_a_frag(X_bf, En, rowBase, kb * 32, lane);
        bf16x16 b = load_b_packed(Wpk_dir, kb, nt, lane);
        acc = __builtin_amdgcn_wmma_f32_16x16x32_bf16(
            false, a, false, b, (short)0, acc, false, false);
    }
    const float bc = bias[col];
    bf16x8 zfrag;
#pragma unroll
    for (int r = 0; r < 8; ++r) zfrag[r] = (__bf16)(acc[r] + bc);
    __bf16* Zp = Zx + (((size_t)dir * 1024 + mt) * NTILES_N + nt) * 256 + lane * 8;
    *(bf16x8*)Zp = zfrag;
}

// ---------------- kernel 5: one fused LSTM step (both dirs) ----------------
// grid: 128 blocks (dir:2 x mt:4 x nt:16), 128 threads (4 waves = 4 gates)
__global__ void k_step(const __bf16* __restrict__ hbf,    // [dir][64][256]
                       float* __restrict__ cstate,        // [dir][64][256]
                       __bf16* __restrict__ hbf_out,      // same as hbf (in-place)
                       const __bf16* __restrict__ Zx,     // fragment layout
                       const __bf16* __restrict__ Wpk,    // swizzled weights
                       float* __restrict__ hfw,           // [t][64][256]
                       float* __restrict__ hbw,           // [t][64][256]
                       int t) {
    __shared__ float zbuf[4][16][16];

    const int lane = threadIdx.x & 31;
    const int g    = threadIdx.x >> 5;         // gate: 0=i 1=j 2=f 3=o
    const int dir  = blockIdx.x >> 6;
    const int rr   = blockIdx.x & 63;
    const int mt   = rr >> 4;
    const int nt   = rr & 15;
    const int t_eff = dir ? (Tn - 1 - t) : t;

    const __bf16* A       = hbf + (size_t)dir * Bn * Hn;       // [64][256]
    const __bf16* Wpk_dir = Wpk + (size_t)dir * KW * G4;       // kblocks 8..15 = Wh
    const int rowBase = mt * 16;
    const int ct      = g * 16 + nt;                           // gate column tile 0..63
    const int n       = lane & 15;

    // Prefetch next step's Zx fragment (gfx1250 global_prefetch_b8):
    // the Zx stream is touched exactly once, so pull step t+1's fragment into
    // cache now to take HBM latency off the sequential recurrence chain.
    const __bf16* ZpBase = Zx + (size_t)dir * 1024 * NTILES_N * 256 + lane * 8;
    if (t + 1 < Tn) {
        int next_teff = dir ? (t_eff - 1) : (t_eff + 1);
        size_t mtile_n = (size_t)next_teff * 4 + mt;
        __builtin_prefetch(ZpBase + (mtile_n * NTILES_N + ct) * 256, 0, 1);
    }

    f32x8 acc = {};
#pragma unroll
    for (int kb = 0; kb < Hn / 32; ++kb) {
        bf16x16 a = load_a_frag(A, Hn, rowBase, kb * 32, lane);
        bf16x16 b = load_b_packed(Wpk_dir, 8 + kb, ct, lane);
        acc = __builtin_amdgcn_wmma_f32_16x16x32_bf16(
            false, a, false, b, (short)0, acc, false, false);
    }

    // add precomputed input-side gate pre-activations (fragment-layout vector load)
    const size_t mtile_g = (size_t)t_eff * 4 + mt;             // time-major tile index
    bf16x8 zfrag = *(const bf16x8*)(ZpBase + (mtile_g * NTILES_N + ct) * 256);
#pragma unroll
    for (int r = 0; r < 8; ++r) {
        int m = ((lane >> 4) << 3) + r;
        zbuf[g][m][n] = acc[r] + (float)zfrag[r];
    }
    __syncthreads();

    // cell update: 128 threads cover 256 (m,n) elements
    for (int e = threadIdx.x; e < 256; e += 128) {
        int m   = e >> 4;
        int nn  = e & 15;
        int bb  = rowBase + m;
        int hid = nt * 16 + nn;
        float zi = zbuf[0][m][nn];
        float zj = zbuf[1][m][nn];
        float zf = zbuf[2][m][nn];
        float zo = zbuf[3][m][nn];
        size_t sidx = (size_t)dir * Bn * Hn + (size_t)bb * Hn + hid;
        float c_old = cstate[sidx];
        float c_new = sigmoidf_(zf + 1.0f) * c_old + sigmoidf_(zi) * tanhf(zj);
        float h     = sigmoidf_(zo) * tanhf(c_new);
        cstate[sidx]  = c_new;
        hbf_out[sidx] = (__bf16)h;
        float* hout = dir ? hbw : hfw;
        hout[(size_t)t_eff * Bn * Hn + (size_t)bb * Hn + hid] = h;
    }
}

// ---------------- kernel 6: readout + xent, per-block partial sums ----------------
__global__ void k_loss_partial(const float* __restrict__ hfw,
                               const float* __restrict__ hbw,
                               const float* __restrict__ W,     // [512][2]
                               const float* __restrict__ b2,    // [2]
                               const float* __restrict__ outs,  // [B][T][2]
                               const float* __restrict__ mask,  // [B][T]
                               float* __restrict__ partial) {
    __shared__ float red[128];
    int idx = blockIdx.x * 128 + threadIdx.x;   // 16384
    int bb  = idx / Tn;
    int t   = idx % Tn;
    float l0 = b2[0], l1 = b2[1];
    const float* hf = hfw + (size_t)t * Bn * Hn + (size_t)bb * Hn;
    const float* hb = hbw + (size_t)t * Bn * Hn + (size_t)bb * Hn;
#pragma unroll 4
    for (int k = 0; k < Hn; ++k) {
        float v = hf[k];
        l0 += v * W[k * 2 + 0];
        l1 += v * W[k * 2 + 1];
    }
#pragma unroll 4
    for (int k = 0; k < Hn; ++k) {
        float v = hb[k];
        l0 += v * W[(Hn + k) * 2 + 0];
        l1 += v * W[(Hn + k) * 2 + 1];
    }
    float mx  = fmaxf(l0, l1);
    float lse = mx + __logf(__expf(l0 - mx) + __expf(l1 - mx));
    float o0  = outs[(size_t)idx * 2 + 0];
    float o1  = outs[(size_t)idx * 2 + 1];
    float xent = -(o0 * (l0 - lse) + o1 * (l1 - lse));
    red[threadIdx.x] = xent * mask[idx];
    __syncthreads();
    for (int s = 64; s > 0; s >>= 1) {
        if (threadIdx.x < s) red[threadIdx.x] += red[threadIdx.x + s];
        __syncthreads();
    }
    if (threadIdx.x == 0) partial[blockIdx.x] = red[0];
}

__global__ void k_loss_final(const float* __restrict__ partial,
                             float* __restrict__ out) {
    __shared__ float red[128];
    red[threadIdx.x] = partial[threadIdx.x];
    __syncthreads();
    for (int s = 64; s > 0; s >>= 1) {
        if (threadIdx.x < s) red[threadIdx.x] += red[threadIdx.x + s];
        __syncthreads();
    }
    if (threadIdx.x == 0) out[0] = red[0] / (float)(Bn * Tn);
}

// ---------------- host launcher ----------------
extern "C" void kernel_launch(void* const* d_in, const int* in_sizes, int n_in,
                              void* d_out, int out_size, void* d_ws, size_t ws_size,
                              hipStream_t stream) {
    const int*   tokens = (const int*)d_in[0];
    const float* outs   = (const float*)d_in[1];
    const float* mask   = (const float*)d_in[2];
    const float* emb    = (const float*)d_in[3];
    const float* Wkf    = (const float*)d_in[4];
    const float* bf     = (const float*)d_in[5];
    const float* Wkb    = (const float*)d_in[6];
    const float* bb     = (const float*)d_in[7];
    const float* W      = (const float*)d_in[8];
    const float* b2     = (const float*)d_in[9];
    float* out = (float*)d_out;

    // workspace carve-up (256B aligned)
    char* ws = (char*)d_ws;
    size_t off = 0;
    auto carve = [&](size_t bytes) -> char* {
        char* p = ws + off;
        off = (off + bytes + 255) & ~(size_t)255;
        return p;
    };
    __bf16* X_bf = (__bf16*)carve((size_t)NROWS * En * 2);        //  8 MB
    __bf16* Wpk  = (__bf16*)carve((size_t)2 * KW * G4 * 2);       //  2 MB
    __bf16* Zx   = (__bf16*)carve((size_t)2 * NROWS * G4 * 2);    // 64 MB
    float*  hfw  = (float*)carve((size_t)Tn * Bn * Hn * 4);       // 16 MB
    float*  hbw  = (float*)carve((size_t)Tn * Bn * Hn * 4);       // 16 MB
    float*  cst  = (float*)carve((size_t)2 * Bn * Hn * 4);
    __bf16* hbf  = (__bf16*)carve((size_t)2 * Bn * Hn * 2);
    float*  part = (float*)carve(128 * 4);

    // 1) weights -> bf16, fragment-swizzled
    k_cvt_weights<<<(2 * KW * G4 + 255) / 256, 256, 0, stream>>>(Wkf, Wkb, Wpk);
    // 2) embedding gather -> bf16 (time-major)
    k_embed<<<(NROWS * En + 255) / 256, 256, 0, stream>>>(tokens, emb, X_bf);
    // 3) zero c / h
    k_init_state<<<(2 * Bn * Hn + 255) / 256, 256, 0, stream>>>(cst, hbf);
    // 4) Zx = X @ Wx + bias, both directions (131072 tiles, 8 waves/block)
    k_zx_gemm<<<131072 / 8, 256, 0, stream>>>(X_bf, Wpk, bf, bb, Zx);
    // 5) sequential recurrence: 256 fused step kernels
    for (int t = 0; t < Tn; ++t) {
        k_step<<<128, 128, 0, stream>>>(hbf, cst, hbf, Zx, Wpk, hfw, hbw, t);
    }
    // 6) loss
    k_loss_partial<<<128, 128, 0, stream>>>(hfw, hbw, W, b2, outs, mask, part);
    k_loss_final<<<1, 128, 0, stream>>>(part, out);
}
